// Net1_10617159155776
// MI455X (gfx1250) — matrix operand order repaired and checked by reading, not compiled, not run on hardware
//
#include <hip/hip_runtime.h>
#include <hip/hip_bf16.h>
#include <stdint.h>

typedef __attribute__((ext_vector_type(16))) __bf16 v16bf;
typedef __attribute__((ext_vector_type(8)))  __bf16 v8bf;
typedef __attribute__((ext_vector_type(8)))  float  v8f;
typedef __attribute__((ext_vector_type(4)))  unsigned int u32x4;
typedef __attribute__((ext_vector_type(8)))  int i32x8;
typedef __attribute__((ext_vector_type(4)))  int i32x4;

#define TM 256          // rows per block (8 waves x 32 rows)
#define TN 128          // cols per block
#define TKK 32          // K slab per TDM stage
#define ABYTES (TM * TKK * 2)              // 16 KB
#define BBYTES (TN * TKK * 2)              // 8 KB
#define BUFBYTES (ABYTES + BBYTES)         // 24 KB
#define EPSV 1e-5f

// ---------------------------------------------------------------------------
// TDM: build D# group0/group1 for a 2D bf16 tile load into LDS (ISA 8.3/8.4)
// 6-arg builtin form (clang-23 / therock headers).
// ---------------------------------------------------------------------------
__device__ __forceinline__ void tdm_load_2d(uint32_t lds_off, const void* gptr,
                                            uint32_t tensor_d0, uint32_t tensor_d1,
                                            uint32_t stride_d0,
                                            uint32_t tile_d0, uint32_t tile_d1)
{
  uint64_t ga = (uint64_t)gptr;
  u32x4 g0;
  g0.x = 1u;                                                  // count=1, user D#
  g0.y = lds_off;                                             // lds_addr (bytes)
  g0.z = (uint32_t)ga;                                        // global_addr[31:0]
  g0.w = (uint32_t)((ga >> 32) & 0x01ffffffu) | (2u << 30);   // [56:32] | type=2
  i32x8 g1;
  g1.s0 = (int)(1u << 16);                                    // data_size=1 (2B), no mask
  g1.s1 = (int)((tensor_d0 & 0xffffu) << 16);                 // tensor_dim0[15:0]
  g1.s2 = (int)((tensor_d0 >> 16) | ((tensor_d1 & 0xffffu) << 16));
  g1.s3 = (int)((tensor_d1 >> 16) | (tile_d0 << 16));         // | tile_dim0
  g1.s4 = (int)(tile_d1 & 0xffffu);                           // tile_dim1, tile_dim2=0
  g1.s5 = (int)stride_d0;                                     // dim0 stride[31:0]
  g1.s6 = 0;                                                  // stride hi, dim1 stride
  g1.s7 = 0;
  i32x4 z4 = {0, 0, 0, 0};
  i32x8 z8 = {0, 0, 0, 0, 0, 0, 0, 0};
  __builtin_amdgcn_tensor_load_to_lds(g0, g1, z4, z4, z8, 0);
}

// ---------------------------------------------------------------------------
// bf16 WMMA GEMM:  C[M,N] = A[M,K] * Bt[N,K]^T (+bias).  K%32==0, N%128==0.
// 8 waves/block; each wave: 32 rows x 128 cols = 16 x wmma_f32_16x16x32_bf16
// per K-slab, sharing one set of 8 B fragments.  TDM double-buffered staging.
// ---------------------------------------------------------------------------
__global__ __launch_bounds__(256) void wmma_gemm_bf16(
    const __bf16* __restrict__ A, const __bf16* __restrict__ Bt,
    float* __restrict__ C, const float* __restrict__ bias,
    int M, int N, int K)
{
  extern __shared__ char smem[];   // [2][A:16KB | B:8KB] = 48KB
  const int m0   = blockIdx.x * TM;
  const int n0   = blockIdx.y * TN;
  const int wave = threadIdx.x >> 5;
  const int lane = threadIdx.x & 31;
  v8f acc[2][8] = {};
  const int nsteps = K / TKK;

  if (threadIdx.x == 0) {
    uint32_t rem = (uint32_t)(M - m0); if (rem > TM) rem = TM;
    tdm_load_2d(0u,      A  + (size_t)m0 * K, (uint32_t)K, rem, (uint32_t)K, TKK, TM);
    tdm_load_2d(ABYTES,  Bt + (size_t)n0 * K, (uint32_t)K, TN,  (uint32_t)K, TKK, TN);
    __builtin_amdgcn_s_wait_tensorcnt(0);
  }
  __syncthreads();

  for (int s = 0; s < nsteps; ++s) {
    const int buf = s & 1;
    if (threadIdx.x == 0 && (s + 1) < nsteps) {             // prefetch next K-slab
      const int k0 = (s + 1) * TKK;
      uint32_t rem = (uint32_t)(M - m0); if (rem > TM) rem = TM;
      uint32_t off = (uint32_t)(buf ^ 1) * BUFBYTES;
      tdm_load_2d(off,          A  + (size_t)m0 * K + k0, (uint32_t)(K - k0), rem, (uint32_t)K, TKK, TM);
      tdm_load_2d(off + ABYTES, Bt + (size_t)n0 * K + k0, (uint32_t)(K - k0), TN,  (uint32_t)K, TKK, TN);
    }
    const __bf16* As = (const __bf16*)(smem + buf * BUFBYTES);
    const __bf16* Bs = (const __bf16*)(smem + buf * BUFBYTES + ABYTES);
    // A fragments: lanes 0-15 -> K 0..7 / 16..23, lanes 16-31 -> K 8..15 / 24..31
    const int r0  = wave * 32 + (lane & 15);
    const int akb = (lane < 16) ? 0 : 8;
    v8bf a0lo = *(const v8bf*)(As + r0 * TKK + akb);
    v8bf a0hi = *(const v8bf*)(As + r0 * TKK + 16 + akb);
    v8bf a1lo = *(const v8bf*)(As + (r0 + 16) * TKK + akb);
    v8bf a1hi = *(const v8bf*)(As + (r0 + 16) * TKK + 16 + akb);
    v16bf a0 = __builtin_shufflevector(a0lo, a0hi, 0,1,2,3,4,5,6,7,8,9,10,11,12,13,14,15);
    v16bf a1 = __builtin_shufflevector(a1lo, a1hi, 0,1,2,3,4,5,6,7,8,9,10,11,12,13,14,15);
    // B fragments: preload all 8 so the 16 WMMAs can issue back-to-back
    const int bkb = (lane < 16) ? 0 : 8;
    v16bf bfr[8];
#pragma unroll
    for (int j = 0; j < 8; ++j) {
      const int bc = j * 16 + (lane & 15);                  // output column (row of Bt)
      v8bf blo = *(const v8bf*)(Bs + bc * TKK + bkb);
      v8bf bhi = *(const v8bf*)(Bs + bc * TKK + 16 + bkb);
      bfr[j] = __builtin_shufflevector(blo, bhi, 0,1,2,3,4,5,6,7,8,9,10,11,12,13,14,15);
    }
#pragma unroll
    for (int j = 0; j < 8; ++j)
      acc[0][j] = __builtin_amdgcn_wmma_f32_16x16x32_bf16(false, a0, false, bfr[j],
                                                          (short)0, acc[0][j], false, false);
#pragma unroll
    for (int j = 0; j < 8; ++j)
      acc[1][j] = __builtin_amdgcn_wmma_f32_16x16x32_bf16(false, a1, false, bfr[j],
                                                          (short)0, acc[1][j], false, false);
    if (threadIdx.x == 0) __builtin_amdgcn_s_wait_tensorcnt(0);
    __syncthreads();
  }

  const int half = lane >> 4;
  const int c16  = lane & 15;
#pragma unroll
  for (int g = 0; g < 2; ++g) {
#pragma unroll
    for (int j = 0; j < 8; ++j) {
      const int col = n0 + j * 16 + c16;
      const float badd = bias ? bias[col] : 0.f;
#pragma unroll
      for (int v = 0; v < 8; ++v) {
        const int row = m0 + wave * 32 + g * 16 + half * 8 + v;
        if (row < M) C[(size_t)row * N + col] = acc[g][j][v] + badd;
      }
    }
  }
}

// ---------------------------------------------------------------------------
// Packing / conversion
// ---------------------------------------------------------------------------
__global__ void pack_w(const float* __restrict__ W, __bf16* __restrict__ Wt,
                       int Fin, int Kpad, int HC)
{
  const int nout = blockIdx.x;                   // W:[Fin,HC] -> Wt:[HC,Kpad] (bf16)
  for (int k = threadIdx.x; k < Kpad; k += blockDim.x)
    Wt[(size_t)nout * Kpad + k] = (k < Fin) ? (__bf16)W[(size_t)k * HC + nout] : (__bf16)0.f;
}

__global__ void cvt_x(const float* __restrict__ X, __bf16* __restrict__ Xb,
                      int Fin, int Kpad)
{
  const int n = blockIdx.x;
  for (int c = threadIdx.x; c < Kpad; c += blockDim.x)
    Xb[(size_t)n * Kpad + c] = (c < Fin) ? (__bf16)X[(size_t)n * Fin + c] : (__bf16)0.f;
}

__global__ void zero_f32(float* p, size_t n)
{ size_t i = (size_t)blockIdx.x * blockDim.x + threadIdx.x; if (i < n) p[i] = 0.f; }
__global__ void zero_u32(unsigned* p, size_t n)
{ size_t i = (size_t)blockIdx.x * blockDim.x + threadIdx.x; if (i < n) p[i] = 0u; }

// ---------------------------------------------------------------------------
// Attention scores: a[n,h] = sum_c h[n,h,c] * att[h,c]
// ---------------------------------------------------------------------------
__global__ __launch_bounds__(256) void att_scores(
    const float* __restrict__ H, const float* __restrict__ att_s,
    const float* __restrict__ att_d, float* __restrict__ as_, float* __restrict__ ad_,
    int C, int HC)
{
  const int n = blockIdx.x, head = threadIdx.x >> 5, lane = threadIdx.x & 31;
  float s = 0.f, d = 0.f;
  for (int c = lane; c < C; c += 32) {
    const float hv = H[(size_t)n * HC + head * C + c];
    s += hv * att_s[head * C + c];
    d += hv * att_d[head * C + c];
  }
  for (int off = 16; off; off >>= 1) { s += __shfl_down(s, off, 32); d += __shfl_down(d, off, 32); }
  if (lane == 0) { as_[n * 8 + head] = s; ad_[n * 8 + head] = d; }
}

// order-preserving f32<->u32 for atomic max
__device__ __forceinline__ unsigned fenc(float x)
{ unsigned b = __float_as_uint(x); return (b & 0x80000000u) ? ~b : (b | 0x80000000u); }
__device__ __forceinline__ float fdec(unsigned u)
{ unsigned b = (u & 0x80000000u) ? (u & 0x7fffffffu) : ~u; return __uint_as_float(b); }

__global__ void edge_max(const float* __restrict__ as_, const float* __restrict__ ad_,
                         const int* __restrict__ ei, int NE, int Nn, unsigned* __restrict__ segm)
{
  const long id = (long)blockIdx.x * blockDim.x + threadIdx.x;
  if (id >= (long)(NE + Nn) * 8) return;
  const int e = (int)(id >> 3), h = (int)(id & 7);
  const int s = (e < NE) ? ei[e] : (e - NE);
  const int d = (e < NE) ? ei[NE + e] : (e - NE);
  float v = as_[s * 8 + h] + ad_[d * 8 + h];
  v = v > 0.f ? v : 0.2f * v;                                // leaky_relu(0.2)
  atomicMax(&segm[d * 8 + h], fenc(v));
}

__global__ void edge_expsum(const float* __restrict__ as_, const float* __restrict__ ad_,
                            const int* __restrict__ ei, int NE, int Nn,
                            const unsigned* __restrict__ segm,
                            float* __restrict__ expv, float* __restrict__ den)
{
  const long id = (long)blockIdx.x * blockDim.x + threadIdx.x;
  if (id >= (long)(NE + Nn) * 8) return;
  const int e = (int)(id >> 3), h = (int)(id & 7);
  const int s = (e < NE) ? ei[e] : (e - NE);
  const int d = (e < NE) ? ei[NE + e] : (e - NE);
  float v = as_[s * 8 + h] + ad_[d * 8 + h];
  v = v > 0.f ? v : 0.2f * v;
  const float ex = __expf(v - fdec(segm[d * 8 + h]));
  expv[(size_t)e * 8 + h] = ex;
  atomicAdd(&den[d * 8 + h], ex);
}

__global__ __launch_bounds__(256) void edge_msg(
    const float* __restrict__ H, const float* __restrict__ expv,
    const float* __restrict__ den, const int* __restrict__ ei, int NE, int Nn,
    float* __restrict__ agg, int HC, int C)
{
  const int e = blockIdx.x;
  const int s = (e < NE) ? ei[e] : (e - NE);
  const int d = (e < NE) ? ei[NE + e] : (e - NE);
  for (int c = threadIdx.x; c < HC; c += blockDim.x) {
    const int h = c / C;
    const float alpha = expv[(size_t)e * 8 + h] / den[d * 8 + h];
    atomicAdd(&agg[(size_t)d * HC + c], H[(size_t)s * HC + c] * alpha);
  }
}

__global__ void bias_relu(float* __restrict__ x, const float* __restrict__ bias, int HC)
{
  const int n = blockIdx.x;
  for (int c = threadIdx.x; c < HC; c += blockDim.x) {
    const float v = x[(size_t)n * HC + c] + bias[c];
    x[(size_t)n * HC + c] = v > 0.f ? v : 0.f;
  }
}

// ---------------------------------------------------------------------------
// BatchNorm (training-mode batch statistics)
// ---------------------------------------------------------------------------
__global__ void bn_stats(const float* __restrict__ x, float* __restrict__ sum,
                         float* __restrict__ sq, int Nn, int HC)
{
  const int r0 = blockIdx.x * 16;
  const int r1 = (r0 + 16 < Nn) ? r0 + 16 : Nn;
  for (int c = threadIdx.x; c < HC; c += blockDim.x) {
    float s = 0.f, q = 0.f;
    for (int r = r0; r < r1; ++r) { const float v = x[(size_t)r * HC + c]; s += v; q += v * v; }
    atomicAdd(&sum[c], s);
    atomicAdd(&sq[c], q);
  }
}

__global__ void bn_apply(const float* __restrict__ a, const float* __restrict__ r,
                         const float* __restrict__ ga, const float* __restrict__ ba,
                         const float* __restrict__ gr, const float* __restrict__ br,
                         const float* __restrict__ st, float fN, int HC,
                         float* __restrict__ y)
{
  const int n = blockIdx.x;
  for (int c = threadIdx.x; c < HC; c += blockDim.x) {
    const float ma = st[c] / fN, va = st[512 + c] / fN - ma * ma;
    float v = (a[(size_t)n * HC + c] - ma) * rsqrtf(va + EPSV) * ga[c] + ba[c];
    if (r) {
      const float mr = st[1024 + c] / fN, vr = st[1536 + c] / fN - mr * mr;
      v += (r[(size_t)n * HC + c] - mr) * rsqrtf(vr + EPSV) * gr[c] + br[c];
    }
    y[(size_t)n * HC + c] = v;
  }
}

// ---------------------------------------------------------------------------
// Pooling + concat
// ---------------------------------------------------------------------------
__global__ void pool_add(const float* __restrict__ x, const int* __restrict__ batch,
                         float* __restrict__ pooled, int HC)
{
  const int n = blockIdx.x, g = batch[n];
  for (int c = threadIdx.x; c < HC; c += blockDim.x)
    atomicAdd(&pooled[(size_t)g * HC + c], x[(size_t)n * HC + c]);
}

__global__ void concat_fc(const float* __restrict__ pooled, const float* __restrict__ gf,
                          float* __restrict__ fcin, int colbase)
{
  const int g = blockIdx.x, c = threadIdx.x;
  if (c < 256)       fcin[g * 552 + colbase + c] = pooled[g * 256 + c];
  else if (c < 276)  fcin[g * 552 + colbase + c] = gf[g * 20 + (c - 256)];
}

// ---------------------------------------------------------------------------
// FC head: single block, 64 graphs, shared-memory staged
// ---------------------------------------------------------------------------
struct FcOff {
  int bn1b, bn1g, bn2b, bn2g, bn3b, bn3g;
  int fc1b, fc1w, fc2b, fc2w, fc3b, fc3w, fc4b, fc4w;
  int r1bnb, r1bng, r1lb, r1lw;
  int r2bnb, r2bng, r2lb, r2lw;
  int r3bnb, r3bng, r3lb, r3lw;
};

__device__ void fc_stage(const float* P, const float* x, int Din,
                         int fcb, int fcw, int bng, int bnb,
                         int rlb, int rlw, int rbng, int rbnb,
                         float* sa, float* sr, float* sy)
{
  const int tid = threadIdx.x;
  for (int idx = tid; idx < 64 * 64; idx += 256) {
    const int g = idx >> 6, o = idx & 63;
    const float* xr = x + g * Din;
    float af = P[fcb + o], ar = P[rlb + o];
    for (int i = 0; i < Din; ++i) {
      const float xv = xr[i];
      af += xv * P[fcw + i * 64 + o];
      ar += xv * P[rlw + i * 64 + o];
    }
    sa[idx] = fmaxf(af, 0.f);
    sr[idx] = ar;
  }
  __syncthreads();
  if (tid < 64) {
    const int o = tid;
    float s1 = 0.f, q1 = 0.f, s2 = 0.f, q2 = 0.f;
    for (int g = 0; g < 64; ++g) {
      const float a = sa[g * 64 + o], r = sr[g * 64 + o];
      s1 += a; q1 += a * a; s2 += r; q2 += r * r;
    }
    const float ma = s1 * (1.f / 64.f), va = q1 * (1.f / 64.f) - ma * ma;
    const float mr = s2 * (1.f / 64.f), vr = q2 * (1.f / 64.f) - mr * mr;
    const float ia = rsqrtf(va + EPSV) * P[bng + o], ir = rsqrtf(vr + EPSV) * P[rbng + o];
    const float ba = P[bnb + o], br = P[rbnb + o];
    for (int g = 0; g < 64; ++g)
      sy[g * 64 + o] = (sa[g * 64 + o] - ma) * ia + ba + (sr[g * 64 + o] - mr) * ir + br;
  }
  __syncthreads();
}

__global__ __launch_bounds__(256) void fc_forward(const float* __restrict__ P, FcOff F,
                                                  const float* __restrict__ xin,
                                                  float* __restrict__ out)
{
  __shared__ float sa[64 * 64], sr[64 * 64], sy[64 * 64];
  fc_stage(P, xin, 552, F.fc1b, F.fc1w, F.bn1g, F.bn1b, F.r1lb, F.r1lw, F.r1bng, F.r1bnb, sa, sr, sy);
  fc_stage(P, sy,  64,  F.fc2b, F.fc2w, F.bn2g, F.bn2b, F.r2lb, F.r2lw, F.r2bng, F.r2bnb, sa, sr, sy);
  fc_stage(P, sy,  64,  F.fc3b, F.fc3w, F.bn3g, F.bn3b, F.r3lb, F.r3lw, F.r3bng, F.r3bnb, sa, sr, sy);
  if (threadIdx.x < 64) {
    const int g = threadIdx.x;
    float acc = P[F.fc4b];
    for (int c = 0; c < 64; ++c) acc += sy[g * 64 + c] * P[F.fc4w + c];
    out[g] = acc;
  }
}

// ===========================================================================
// Host side
// ===========================================================================
struct BnO  { long b, g; };
struct LinO { long b, w; };
struct ConvO { long ad, as, bias, w; };
struct ResO { BnO bn; LinO lin; };
struct GatO { BnO bn1, bn2, bn3, bn4; ConvO c1, c2, c3, c4; ResO r1, r2, r3; };
struct FcO  { BnO bn1, bn2, bn3; LinO fc1, fc2, fc3, fc4; ResO r1, r2, r3; };

extern "C" void kernel_launch(void* const* d_in, const int* in_sizes, int n_in,
                              void* d_out, int out_size, void* d_ws, size_t ws_size,
                              hipStream_t stream)
{
  const float* P    = (const float*)d_in[0];
  const float* sx   = (const float*)d_in[1];
  const float* sgf  = (const float*)d_in[2];
  const float* vx   = (const float*)d_in[3];
  const float* vgf  = (const float*)d_in[4];
  const int*   sei  = (const int*)d_in[5];
  const int*   sbt  = (const int*)d_in[6];
  const int*   vei  = (const int*)d_in[7];
  const int*   vbt  = (const int*)d_in[8];

  const int Nn = in_sizes[1] / 38;
  const int NE = in_sizes[5] / 2;
  const int Et = NE + Nn;                       // with self loops

  // ---- parameter offsets (jax dict-sorted flatten order) -------------------
  long o = 0;
  auto take     = [&](long n) { long r = o; o += n; return r; };
  auto takeBn   = [&](int d)  { BnO b; b.b = take(d); b.g = take(d); return b; };
  auto takeLin  = [&](int i, int out_) { LinO l; l.b = take(out_); l.w = take((long)i * out_); return l; };
  auto takeConv = [&](int i, int c) { ConvO cv; cv.ad = take(8 * c); cv.as = take(8 * c);
                                      cv.bias = take(8 * c); cv.w = take((long)i * 8 * c); return cv; };
  auto takeRes  = [&](int i, int d) { ResO r; r.bn = takeBn(d); r.lin = takeLin(i, d); return r; };
  FcO fc;
  fc.bn1 = takeBn(64); fc.bn2 = takeBn(64); fc.bn3 = takeBn(64);
  fc.fc1 = takeLin(552, 64); fc.fc2 = takeLin(64, 64); fc.fc3 = takeLin(64, 64); fc.fc4 = takeLin(64, 1);
  fc.r1 = takeRes(552, 64); fc.r2 = takeRes(64, 64); fc.r3 = takeRes(64, 64);
  auto takeGat = [&]() {
    GatO g;
    g.bn1 = takeBn(512); g.bn2 = takeBn(512); g.bn3 = takeBn(512); g.bn4 = takeBn(256);
    g.c1 = takeConv(38, 64); g.c2 = takeConv(512, 64); g.c3 = takeConv(512, 64); g.c4 = takeConv(512, 32);
    g.r1 = takeRes(38, 512); g.r2 = takeRes(512, 512); g.r3 = takeRes(512, 512);
    return g;
  };
  GatO g1 = takeGat(), g2 = takeGat();

  // ---- workspace layout ----------------------------------------------------
  char* ws = (char*)d_ws;
  size_t wo = 0;
  auto walloc = [&](size_t bytes) { size_t r = wo; wo = (wo + bytes + 255) & ~(size_t)255; return r; };
  float*  XCUR  = (float*)(ws + walloc((size_t)Nn * 512 * 4));
  __bf16* XB    = (__bf16*)(ws + walloc((size_t)Nn * 512 * 2));
  float*  Hb    = (float*)(ws + walloc((size_t)Nn * 512 * 4));
  float*  Rb    = (float*)(ws + walloc((size_t)Nn * 512 * 4));
  __bf16* WT    = (__bf16*)(ws + walloc((size_t)512 * 512 * 2));
  float*  ASRC  = (float*)(ws + walloc((size_t)Nn * 8 * 4));
  float*  ADST  = (float*)(ws + walloc((size_t)Nn * 8 * 4));
  unsigned* SEGM = (unsigned*)(ws + walloc((size_t)Nn * 8 * 4));
  float*  DEN   = (float*)(ws + walloc((size_t)Nn * 8 * 4));
  float*  EXPV  = (float*)(ws + walloc((size_t)Et * 8 * 4));
  float*  STATS = (float*)(ws + walloc(2048 * 4));
  float*  POOL  = (float*)(ws + walloc((size_t)64 * 256 * 4));
  float*  FCIN  = (float*)(ws + walloc((size_t)64 * 552 * 4));

  auto gz = [](size_t n) { return (unsigned)((n + 255) / 256); };
  const int mt = (Nn + TM - 1) / TM;

  auto run_gat = [&](const GatO& g, const float* X0, const float* gf,
                     const int* ei, const int* batch, int colbase) {
    struct Blk { const ConvO* cv; const BnO* bn; const ResO* rs; int Fin, Kpad, HC, C; };
    Blk blks[4] = {
      { &g.c1, &g.bn1, &g.r1, 38,  64,  512, 64 },
      { &g.c2, &g.bn2, &g.r2, 512, 512, 512, 64 },
      { &g.c3, &g.bn3, &g.r3, 512, 512, 512, 64 },
      { &g.c4, &g.bn4, nullptr, 512, 512, 256, 32 },
    };
    const float* xsrc = X0;
    for (int b = 0; b < 4; ++b) {
      const Blk& B = blks[b];
      float* AGG = XCUR;   // input already converted into XB; reuse as aggregator
      cvt_x<<<Nn, 256, 0, stream>>>(xsrc, XB, B.Fin, B.Kpad);
      pack_w<<<B.HC, 256, 0, stream>>>(P + B.cv->w, WT, B.Fin, B.Kpad, B.HC);
      wmma_gemm_bf16<<<dim3(mt, B.HC / 128), 256, 2 * BUFBYTES, stream>>>(XB, WT, Hb, nullptr, Nn, B.HC, B.Kpad);
      if (B.rs) {
        pack_w<<<B.HC, 256, 0, stream>>>(P + B.rs->lin.w, WT, B.Fin, B.Kpad, B.HC);
        wmma_gemm_bf16<<<dim3(mt, B.HC / 128), 256, 2 * BUFBYTES, stream>>>(XB, WT, Rb, P + B.rs->lin.b, Nn, B.HC, B.Kpad);
      }
      att_scores<<<Nn, 256, 0, stream>>>(Hb, P + B.cv->as, P + B.cv->ad, ASRC, ADST, B.C, B.HC);
      zero_u32<<<gz((size_t)Nn * 8), 256, 0, stream>>>(SEGM, (size_t)Nn * 8);
      zero_f32<<<gz((size_t)Nn * 8), 256, 0, stream>>>(DEN, (size_t)Nn * 8);
      zero_f32<<<gz((size_t)Nn * B.HC), 256, 0, stream>>>(AGG, (size_t)Nn * B.HC);
      const long tot = (long)Et * 8;
      edge_max<<<(unsigned)((tot + 255) / 256), 256, 0, stream>>>(ASRC, ADST, ei, NE, Nn, SEGM);
      edge_expsum<<<(unsigned)((tot + 255) / 256), 256, 0, stream>>>(ASRC, ADST, ei, NE, Nn, SEGM, EXPV, DEN);
      edge_msg<<<Et, 256, 0, stream>>>(Hb, EXPV, DEN, ei, NE, Nn, AGG, B.HC, B.C);
      bias_relu<<<Nn, 256, 0, stream>>>(AGG, P + B.cv->bias, B.HC);
      zero_f32<<<8, 256, 0, stream>>>(STATS, 2048);
      bn_stats<<<(Nn + 15) / 16, 256, 0, stream>>>(AGG, STATS, STATS + 512, Nn, B.HC);
      if (B.rs) bn_stats<<<(Nn + 15) / 16, 256, 0, stream>>>(Rb, STATS + 1024, STATS + 1536, Nn, B.HC);
      bn_apply<<<Nn, 256, 0, stream>>>(AGG, B.rs ? Rb : nullptr,
          P + B.bn->g, P + B.bn->b,
          B.rs ? P + B.rs->bn.g : P, B.rs ? P + B.rs->bn.b : P,
          STATS, (float)Nn, B.HC, XCUR);
      xsrc = XCUR;
    }
    zero_f32<<<64, 256, 0, stream>>>(POOL, (size_t)64 * 256);
    pool_add<<<Nn, 256, 0, stream>>>(XCUR, batch, POOL, 256);
    concat_fc<<<64, 288, 0, stream>>>(POOL, gf, FCIN, colbase);
  };

  run_gat(g1, sx, sgf, sei, sbt, 0);
  run_gat(g2, vx, vgf, vei, vbt, 276);

  FcOff F;
  F.bn1b = (int)fc.bn1.b;  F.bn1g = (int)fc.bn1.g;
  F.bn2b = (int)fc.bn2.b;  F.bn2g = (int)fc.bn2.g;
  F.bn3b = (int)fc.bn3.b;  F.bn3g = (int)fc.bn3.g;
  F.fc1b = (int)fc.fc1.b;  F.fc1w = (int)fc.fc1.w;
  F.fc2b = (int)fc.fc2.b;  F.fc2w = (int)fc.fc2.w;
  F.fc3b = (int)fc.fc3.b;  F.fc3w = (int)fc.fc3.w;
  F.fc4b = (int)fc.fc4.b;  F.fc4w = (int)fc.fc4.w;
  F.r1bnb = (int)fc.r1.bn.b; F.r1bng = (int)fc.r1.bn.g; F.r1lb = (int)fc.r1.lin.b; F.r1lw = (int)fc.r1.lin.w;
  F.r2bnb = (int)fc.r2.bn.b; F.r2bng = (int)fc.r2.bn.g; F.r2lb = (int)fc.r2.lin.b; F.r2lw = (int)fc.r2.lin.w;
  F.r3bnb = (int)fc.r3.bn.b; F.r3bng = (int)fc.r3.bn.g; F.r3lb = (int)fc.r3.lin.b; F.r3lw = (int)fc.r3.lin.w;
  fc_forward<<<1, 256, 0, stream>>>(P, F, FCIN, (float*)d_out);
}
